// My_gpu_pseudo_lesion_adder_10333691314543
// MI455X (gfx1250) — matrix operand order repaired and checked by reading, not compiled, not run on hardware
//
#include <hip/hip_runtime.h>
#include <hip/hip_bf16.h>
#include <math.h>

// ---------------------------------------------------------------------------
// MI455X (gfx1250) implementation of the pseudo-lesion adder.
//
// Memory-bound (~650 MB traffic -> ~28us at 23.3 TB/s). All Gaussian blurs
// act on {0,1} masks -> masks kept in f16 (halves blur traffic). Each 1-D
// 9-tap conv runs as a 16x16x32 WMMA GEMM (A = halo strip, B = constant band
// matrix B[k][n]=g[k-n], f32 accumulate). A workgroup (4 waves) computes a
// 16-row x 64-output strip whose 16x72 halo is staged into LDS once by the
// Tensor Data Mover (tensor_load_to_lds + s_wait_tensorcnt), with TDM
// zero-fill providing right-edge padding.
// ---------------------------------------------------------------------------

typedef __attribute__((ext_vector_type(16))) _Float16 v16h;
typedef __attribute__((ext_vector_type(8)))  float    v8f;
typedef __attribute__((ext_vector_type(4)))  unsigned u32x4;
typedef __attribute__((ext_vector_type(8)))  int      i32x8;
typedef __attribute__((ext_vector_type(4)))  int      i32x4;

#if defined(__has_builtin)
#if __has_builtin(__builtin_amdgcn_tensor_load_to_lds)
#define HAVE_TDM 1
#endif
#endif
#ifndef HAVE_TDM
#define HAVE_TDM 0
#endif

#define THRESH 0.07f
#define NCAND  2048

// ---------------- deterministic counter-based RNG (surrogate for jax PRNG) --
__device__ __forceinline__ unsigned hash_u32(unsigned x, unsigned seed) {
  x ^= seed;
  x *= 0x9E3779B1u; x ^= x >> 16;
  x *= 0x85EBCA77u; x ^= x >> 13;
  x *= 0xC2B2AE3Du; x ^= x >> 16;
  return x;
}
__device__ __forceinline__ float uniform01(unsigned h) {
  return (float)(h >> 8) * (1.0f / 16777216.0f);
}
__device__ __forceinline__ float gaussianf(unsigned idx, unsigned seed) {
  unsigned h1 = hash_u32(idx, seed);
  unsigned h2 = hash_u32(idx, seed ^ 0xA511E9B3u);
  float u1 = fmaxf(uniform01(h1), 1.1754944e-38f);
  float u2 = uniform01(h2);
  return sqrtf(-2.0f * logf(u1)) * cosf(6.28318530718f * u2);
}

// ---------------- mask init: (target_curr > 0) as f16 0/1 -------------------
__global__ void init_mask(const float* __restrict__ tgt, _Float16* __restrict__ m, int V) {
  int i = blockIdx.x * blockDim.x + threadIdx.x;
  int stride = gridDim.x * blockDim.x;
  for (; i < V; i += stride) m[i] = (_Float16)((tgt[i] > 0.0f) ? 1.0f : 0.0f);
}

// ---------------- WMMA 1-D Gaussian blur pass -------------------------------
// Block = 128 threads (4 waves). Block computes 16 rows x 64 conv outputs;
// wave wv handles conv columns [blockIdx.x*64 + wv*16, +16).
// LDS staging strip: 16 x span (span = 72 - left_clamp) f16 elements.
//   convContig (W pass): layout [row][cofs], pitch = span.
//   else (H/D passes):   layout [cofs][row], pitch = 16 (TDM line = 16 elems).
// Fragments per CDNA5 ISA 7.12.2:
//   A 16x32 f16: lane L: M=L&15; halves e<8 -> K=e+hi*8, e>=8 -> K=16+(e-8)+hi*8.
//   B 32x16 f16: half e -> K same as A, N=lane&15. C/D f32: vgpr v: M=v+8*hi, N=lane&15.
__global__ __launch_bounds__(128) void blur_pass(
    const _Float16* __restrict__ src, _Float16* __restrict__ dst,
    int convStride, int convLen, int rowStride, long mulZ, int thresholdOut)
{
  __shared__ __align__(16) char smem[72 * 16 * 2 + 4 * 16 * 17 * 4];
  _Float16* stg = (_Float16*)smem;                    // staging strip at LDS offset 0

  const int tid  = threadIdx.x;
  const int wv   = tid >> 5;
  const int lane = tid & 31;
  const int m    = lane & 15;
  const int hi   = lane >> 4;
  const int n    = lane & 15;
  const int convBlock = blockIdx.x * 64;
  const int convBase  = convBlock + wv * 16;
  const long base = (long)blockIdx.z * mulZ + (long)blockIdx.y * 16 * rowStride;
  const int convContig = (convStride == 1);

  const int cLo   = convBlock - 4;                    // halo start (may be <0)
  const int shift = (cLo < 0) ? -cLo : 0;
  const int cs    = cLo + shift;                      // clamped window start
  const int span  = 72 - shift;                       // staged elements along conv
  const int avail = convLen - cs;                     // real elements; rest zero-fill

#if HAVE_TDM
  if (tid < 32) {                                     // one wave issues the DMA
    unsigned long long ga = (unsigned long long)(src + base + (long)cs * convStride);
    // D# group0: count=1 | lds_addr=0 | global_addr | type=2
    u32x4 g0 = { 1u, 0u, (unsigned)(ga & 0xFFFFFFFFull),
                 (unsigned)((ga >> 32) & 0x1FFFFFFull) | (2u << 30) };
    unsigned dim0, dim1, td0, td1; unsigned long long str0;
    if (convContig) { dim0 = (unsigned)avail; td0 = (unsigned)span;
                      dim1 = 16u;             td1 = 16u;
                      str0 = (unsigned long long)rowStride; }
    else            { dim0 = 16u;             td0 = 16u;
                      dim1 = (unsigned)avail; td1 = (unsigned)span;
                      str0 = (unsigned long long)convStride; }
    // D# group1 bit-packing (ISA 8.4): data_size=1 (2B), masks/pad/iterate = 0
    i32x8 g1;
    g1[0] = (int)(1u << 16);                               // data_size
    g1[1] = (int)((dim0 & 0xFFFFu) << 16);                 // tensor_dim0[15:0] @63:48
    g1[2] = (int)((dim0 >> 16) | ((dim1 & 0xFFFFu) << 16));// dim0[31:16] | dim1[15:0]
    g1[3] = (int)((dim1 >> 16) | (td0 << 16));             // dim1[31:16] | tile_dim0
    g1[4] = (int)td1;                                      // tile_dim1, tile_dim2=0
    g1[5] = (int)(unsigned)(str0 & 0xFFFFFFFFull);         // dim0_stride[31:0]
    g1[6] = (int)(unsigned)((str0 >> 32) & 0xFFFFull);     // dim0_stride[47:32]
    g1[7] = 0;                                             // dim1_stride (unused, 2D)
    i32x4 g2 = { 1, 0, 0, 0 };                             // tensor_dim2=1, rest 0
    i32x4 g3 = { 0, 0, 0, 0 };
    i32x8 g4 = { 0, 0, 0, 0, 0, 0, 0, 0 };                 // extra operand (clang-23 form)
    // amdgpu-toolchain (clang-23) 6-arg form: (g0, g1, g2, g3, g4, cpol)
    __builtin_amdgcn_tensor_load_to_lds(g0, g1, g2, g3, g4, 0);
    __builtin_amdgcn_s_wait_tensorcnt(0);
  }
  __syncthreads();
#else
  // Fallback: cooperative coalesced staging.
  {
    int total = span * 16;
    for (int t = tid; t < total; t += 128) {
      int row, cofs;
      if (convContig) { row = t / span; cofs = t - row * span; }
      else            { row = t & 15;   cofs = t >> 4; }
      int c = cs + cofs;
      float v = 0.0f;
      if (c < convLen) v = (float)src[base + (long)row * rowStride + (long)c * convStride];
      stg[t] = (_Float16)v;
    }
  }
  __syncthreads();
#endif

  // 9-tap MONAI erf kernel, sigma=1, truncated=4 (clamped at 0)
  float g[9];
#pragma unroll
  for (int t = 0; t < 9; ++t) {
    float x = (float)(t - 4);
    const float s = 1.41421356237f;                   // sigma*sqrt(2)
    float v = 0.5f * (erff((x + 0.5f) / s) - erff((x - 0.5f) / s));
    g[t] = v < 0.0f ? 0.0f : v;
  }

  // A fragment from LDS (K in [0,24) used; right edge already zero-filled)
  v16h a;
#pragma unroll
  for (int e = 0; e < 16; ++e) {
    int K = (e < 8) ? (e + hi * 8) : (16 + (e - 8) + hi * 8);
    int c = convBase + K - 4;
    float val = 0.0f;
    if (K < 24 && c >= 0) {
      int ci = c - cs;
      int idx = convContig ? (m * span + ci) : (ci * 16 + m);
      val = (float)stg[idx];
    }
    a[e] = (_Float16)val;
  }

  // B fragment: constant band matrix B[K][n] = g[K-n] for 0<=K-n<=8.
  v16h b;
#pragma unroll
  for (int e = 0; e < 16; ++e) {
    int K = (e < 8) ? (e + hi * 8) : (16 + (e - 8) + hi * 8);
    int d = K - n;
    b[e] = (_Float16)((d >= 0 && d <= 8) ? g[d] : 0.0f);
  }

  v8f c8 = {};
  c8 = __builtin_amdgcn_wmma_f32_16x16x32_f16(
      /*neg_a=*/false, a, /*neg_b=*/false, b,
      /*c_mod=*/(short)0, c8, /*reuse_a=*/false, /*reuse_b=*/false);

  if (rowStride == 1) {
    // H/D passes: M is the unit-stride dim -> per-wave LDS transpose so the
    // global stores are coalesced along M.
    float (*tw)[17] = (float (*)[17])(smem + 72 * 16 * 2 + wv * 16 * 17 * 4);
#pragma unroll
    for (int v = 0; v < 8; ++v) tw[v + hi * 8][n] = c8[v];
    __syncthreads();
#pragma unroll
    for (int i = 0; i < 8; ++i) {
      int nn = 2 * i + hi;
      float val = tw[m][nn];
      if (thresholdOut) val = (val > THRESH) ? 1.0f : 0.0f;
      dst[base + m + (long)(convBase + nn) * convStride] = (_Float16)val;
    }
  } else {
    // W pass: N already unit-stride -> direct coalesced stores.
#pragma unroll
    for (int v = 0; v < 8; ++v) {
      float val = c8[v];
      if (thresholdOut) val = (val > THRESH) ? 1.0f : 0.0f;
      dst[base + (long)(v + hi * 8) * rowStride + (convBase + n)] = (_Float16)val;
    }
  }
}

// ---------------- candidate filter for top-20 seed selection ----------------
__global__ void seed_candidates(const _Float16* __restrict__ tgtBig, _Float16* __restrict__ seeds,
                                unsigned* __restrict__ candCount,
                                unsigned* __restrict__ candScore, unsigned* __restrict__ candIdx,
                                int V, unsigned cutoff)
{
  int i = blockIdx.x * blockDim.x + threadIdx.x;
  int stride = gridDim.x * blockDim.x;
  for (; i < V; i += stride) {
    seeds[i] = (_Float16)0.0f;                    // clear seed volume in same pass
    if ((float)tgtBig[i] == 0.0f) {               // outside forbidden region
      unsigned h = hash_u32((unsigned)i, 0xC0FFEE01u);
      if (h >= cutoff) {                          // ~128 expected survivors
        unsigned slot = atomicAdd(candCount, 1u);
        if (slot < NCAND) { candScore[slot] = h; candIdx[slot] = (unsigned)i; }
      }
    }
  }
}

// Exact top-20 over candidates; tie-break (score, then smaller idx) so the
// result is independent of the atomicAdd slot order -> deterministic replays.
__global__ void select_seeds(const unsigned* __restrict__ candCount,
                             const unsigned* __restrict__ candScore,
                             const unsigned* __restrict__ candIdx,
                             _Float16* __restrict__ seeds)
{
  __shared__ unsigned long long skey[NCAND];
  __shared__ unsigned long long red[256];
  unsigned cnt = *candCount; if (cnt > (unsigned)NCAND) cnt = NCAND;
  for (unsigned i = threadIdx.x; i < (unsigned)NCAND; i += 256u)
    skey[i] = (i < cnt)
      ? ((((unsigned long long)candScore[i]) << 32) | (unsigned)(~candIdx[i]))
      : 0ull;
  __syncthreads();
  for (int k = 0; k < 20; ++k) {
    unsigned long long best = 0ull;
    for (unsigned i = threadIdx.x; i < (unsigned)NCAND; i += 256u)
      if (skey[i] > best) best = skey[i];
    red[threadIdx.x] = best;
    __syncthreads();
    for (int s = 128; s > 0; s >>= 1) {
      if ((int)threadIdx.x < s && red[threadIdx.x + s] > red[threadIdx.x])
        red[threadIdx.x] = red[threadIdx.x + s];
      __syncthreads();
    }
    unsigned long long win = red[0];
    __syncthreads();
    if (win != 0ull) {
      for (unsigned i = threadIdx.x; i < (unsigned)NCAND; i += 256u)
        if (skey[i] == win) {
          skey[i] = 0ull;
          unsigned idx = ~(unsigned)(win & 0xFFFFFFFFull);
          seeds[idx] = (_Float16)1.0f;
        }
    }
    __syncthreads();
  }
}

// ---------------- fused min/max over the 4 standardized fields --------------
#define SEED_ADC 0x1B873593u
#define SEED_HBV 0x7FEB352Du

__global__ void minmax_partial(const float* __restrict__ data, const _Float16* __restrict__ R,
                               float* __restrict__ partial, int V)
{
  const float* d0 = data;
  const float* d1 = data + V;
  float mn[4] = { INFINITY,  INFINITY,  INFINITY,  INFINITY};
  float mx[4] = {-INFINITY, -INFINITY, -INFINITY, -INFINITY};
  int i = blockIdx.x * blockDim.x + threadIdx.x;
  int stride = gridDim.x * blockDim.x;
  for (; i < V; i += stride) {
    bool inR = ((float)R[i] != 0.0f);
    float f[4];
    f[0] = inR ? 0.0f : d0[i];
    f[1] = inR ? 0.0f : d1[i];
    f[2] = inR ? (0.2f + 0.1f * gaussianf((unsigned)i, SEED_ADC)) : 0.0f;
    f[3] = inR ? (0.8f + 0.1f * gaussianf((unsigned)i, SEED_HBV)) : 0.0f;
#pragma unroll
    for (int j = 0; j < 4; ++j) { mn[j] = fminf(mn[j], f[j]); mx[j] = fmaxf(mx[j], f[j]); }
  }
  __shared__ float smn[256][4], smx[256][4];
#pragma unroll
  for (int j = 0; j < 4; ++j) { smn[threadIdx.x][j] = mn[j]; smx[threadIdx.x][j] = mx[j]; }
  __syncthreads();
  for (int s = 128; s > 0; s >>= 1) {
    if ((int)threadIdx.x < s)
#pragma unroll
      for (int j = 0; j < 4; ++j) {
        smn[threadIdx.x][j] = fminf(smn[threadIdx.x][j], smn[threadIdx.x + s][j]);
        smx[threadIdx.x][j] = fmaxf(smx[threadIdx.x][j], smx[threadIdx.x + s][j]);
      }
    __syncthreads();
  }
  if (threadIdx.x == 0)
#pragma unroll
    for (int j = 0; j < 4; ++j) {
      partial[blockIdx.x * 8 + 2 * j]     = smn[0][j];
      partial[blockIdx.x * 8 + 2 * j + 1] = smx[0][j];
    }
}

__global__ void minmax_final(const float* __restrict__ partial, int nblk,
                             float* __restrict__ minmax)
{
  __shared__ float smn[256][4], smx[256][4];
  float mn[4] = { INFINITY,  INFINITY,  INFINITY,  INFINITY};
  float mx[4] = {-INFINITY, -INFINITY, -INFINITY, -INFINITY};
  for (int b = threadIdx.x; b < nblk; b += 256)
#pragma unroll
    for (int j = 0; j < 4; ++j) {
      mn[j] = fminf(mn[j], partial[b * 8 + 2 * j]);
      mx[j] = fmaxf(mx[j], partial[b * 8 + 2 * j + 1]);
    }
#pragma unroll
  for (int j = 0; j < 4; ++j) { smn[threadIdx.x][j] = mn[j]; smx[threadIdx.x][j] = mx[j]; }
  __syncthreads();
  for (int s = 128; s > 0; s >>= 1) {
    if ((int)threadIdx.x < s)
#pragma unroll
      for (int j = 0; j < 4; ++j) {
        smn[threadIdx.x][j] = fminf(smn[threadIdx.x][j], smn[threadIdx.x + s][j]);
        smx[threadIdx.x][j] = fmaxf(smx[threadIdx.x][j], smx[threadIdx.x + s][j]);
      }
    __syncthreads();
  }
  if (threadIdx.x == 0)
#pragma unroll
    for (int j = 0; j < 4; ++j) { minmax[2 * j] = smn[0][j]; minmax[2 * j + 1] = smx[0][j]; }
}

// ---------------- final composition: channels 0 and 1 -----------------------
__device__ __forceinline__ float stdv(float x, float mn, float mx) {
  float den = mx - mn;                   // standardd + nan_to_num(0/0) -> 0
  return den > 0.0f ? (x - mn) / den : 0.0f;
}

__global__ void compose(const float* __restrict__ data, const _Float16* __restrict__ R,
                        const float* __restrict__ minmax, float* __restrict__ out, int V)
{
  __shared__ float mm[8];
  if (threadIdx.x < 8) mm[threadIdx.x] = minmax[threadIdx.x];
  __syncthreads();
  const float* d0 = data;
  const float* d1 = data + V;
  int i = blockIdx.x * blockDim.x + threadIdx.x;
  int stride = gridDim.x * blockDim.x;
  for (; i < V; i += stride) {
    bool inR = ((float)R[i] != 0.0f);
    float a0 = inR ? 0.0f : d0[i];
    float a1 = inR ? 0.0f : d1[i];
    float nA = inR ? (0.2f + 0.1f * gaussianf((unsigned)i, SEED_ADC)) : 0.0f;
    float nH = inR ? (0.8f + 0.1f * gaussianf((unsigned)i, SEED_HBV)) : 0.0f;
    out[i]     = stdv(a0, mm[0], mm[1]) + stdv(nA, mm[4], mm[5]);
    out[V + i] = stdv(a1, mm[2], mm[3]) + stdv(nH, mm[6], mm[7]);
  }
}

// ---------------------------------------------------------------------------
extern "C" void kernel_launch(void* const* d_in, const int* in_sizes, int n_in,
                              void* d_out, int out_size, void* d_ws, size_t ws_size,
                              hipStream_t stream)
{
  (void)in_sizes; (void)n_in; (void)out_size; (void)ws_size;
  const float* data = (const float*)d_in[0];     // [1,5,128,256,256]
  const float* tgt  = (const float*)d_in[1];     // [1,128,256,256]
  float* out = (float*)d_out;                    // [1,5,128,256,256]

  const int D = 128, H = 256, W = 256;
  const int V = D * H * W;                       // 8,388,608

  // workspace layout: two f16 mask volumes + small scratch
  char* ws = (char*)d_ws;
  _Float16* bufM = (_Float16*)ws;
  _Float16* bufN = (_Float16*)(ws + (size_t)V * sizeof(_Float16));
  float*    partial   = (float*)(ws + (size_t)2 * V * sizeof(_Float16));
  float*    minmax    = partial + 2048 * 8;
  unsigned* candCount = (unsigned*)(minmax + 8);
  unsigned* candScore = candCount + 1;
  unsigned* candIdx   = candScore + NCAND;

  (void)hipMemsetAsync(candCount, 0, sizeof(unsigned), stream);

  auto blurW = [&](const _Float16* s, _Float16* d, int th) {
    blur_pass<<<dim3(W / 64, H / 16, D), 128, 0, stream>>>(s, d, 1, W, W, (long)H * W, th);
  };
  auto blurH = [&](const _Float16* s, _Float16* d, int th) {
    blur_pass<<<dim3(H / 64, W / 16, D), 128, 0, stream>>>(s, d, W, H, 1, (long)H * W, th);
  };
  auto blurD = [&](const _Float16* s, _Float16* d, int th) {
    blur_pass<<<dim3(D / 64, W / 16, H), 128, 0, stream>>>(s, d, H * W, D, 1, (long)W, th);
  };

  // 1) forbidden region: dilate(tgt>0, 3) = 2 rounds of blur3 -> >0.07
  init_mask<<<2048, 256, 0, stream>>>(tgt, bufM, V);
  blurW(bufM, bufN, 0); blurH(bufN, bufM, 0); blurD(bufM, bufN, 1);   // round 1 -> bufN
  blurW(bufN, bufM, 0); blurH(bufM, bufN, 0); blurD(bufN, bufM, 1);   // round 2 -> bufM

  // 2) pick 20 seeds outside target_big (deterministic hash-uniform top-20)
  seed_candidates<<<2048, 256, 0, stream>>>(bufM, bufN, candCount, candScore, candIdx,
                                            V, 0xFFFF0000u);
  select_seeds<<<1, 256, 0, stream>>>(candCount, candScore, candIdx, bufN);

  // 3) res_bool = dilate(seeds, 2)  (the b & ~(~a & b) term reduces to a)
  blurW(bufN, bufM, 0); blurH(bufM, bufN, 0); blurD(bufN, bufM, 1);   // -> bufM

  // 4) standardize + compose channels 0/1; copy channels 2..4 untouched
  minmax_partial<<<2048, 256, 0, stream>>>(data, bufM, partial, V);
  minmax_final<<<1, 256, 0, stream>>>(partial, 2048, minmax);
  compose<<<2048, 256, 0, stream>>>(data, bufM, minmax, out, V);
  (void)hipMemcpyAsync(out + (size_t)2 * V, data + (size_t)2 * V,
                       (size_t)3 * V * sizeof(float), hipMemcpyDeviceToDevice, stream);
}
